// PinnLayer_50938312131078
// MI455X (gfx1250) — matrix operand order repaired and checked by reading, not compiled, not run on hardware
//
#include <hip/hip_runtime.h>
#include <hip/hip_bf16.h>

// ---------------- problem constants ----------------
#define N2BUS   600     // 2*N_BUS
#define N2G     138     // 2*N_GBUS
#define NLINE   411
#define BATCH_N 32768
#define BATCH_F 32768.0f

typedef __attribute__((ext_vector_type(16))) __bf16       v16bf;
typedef __attribute__((ext_vector_type(8)))  float        v8f;
typedef __attribute__((ext_vector_type(4)))  unsigned int u32x4;

// LDS layout strides (bf16 elements), picked for 16B alignment + conflict-free b128 loads
#define A_STRIDE 616    // >= 608 padded-K, 1232B row (16B aligned, 308 dwords -> distinct banks)
#define B_ROWP   40     // 32 used, 80B row (16B aligned, 20 dwords -> distinct banks)

// Packed weight panels in workspace (row-major [n][k] bf16, hi & lo)
#define WV_ROWS  2048   // 600(Y) + 600(Yconj) + 822(Ibr pairs) padded to 128 tiles
#define WV_K     608    // K=600 padded to 19*32
#define WP_ROWS  640    // Map_L^T 600 rows padded
#define WP_K     608
#define WGS_ROWS 256    // Map_g 138 rows padded to 16 tiles (one tg sweep = 8 tiles)
#define WGS_K    608
#define WGEN_ROWS 640   // Map_g^T 600 rows padded
#define WGEN_K   160    // K=138 padded to 5*32
#define NUM_WG   1024   // 32768 / 32 rows per WG

__device__ __forceinline__ void split_bf16(float x, __bf16& h, __bf16& l) {
  h = (__bf16)x;
  l = (__bf16)(x - (float)h);
}

// ---------------- weight packing kernels ----------------
// Wv rows: [0,600)=Y[n][k]; [600,1200)=Yconj; [1200,2022)=interleaved (re,im) rows of M=Ybr@IM
__global__ void pack_wv(const float* __restrict__ Y, const float* __restrict__ Yc,
                        const float* __restrict__ Ybr, const float* __restrict__ IM,
                        __bf16* __restrict__ Wh, __bf16* __restrict__ Wl) {
  const int row = blockIdx.x;
  for (int k = threadIdx.x; k < WV_K; k += 256) {
    float v = 0.f;
    if (k < 600) {
      if (row < 600)       v = Y[row * 600 + k];
      else if (row < 1200) v = Yc[(row - 600) * 600 + k];
      else {
        int p = row - 1200;
        if (p < 822) {
          int l = (p >> 1) + 411 * (p & 1);   // even->re row t, odd->im row t+411
          float s = 0.f;
          for (int j = 0; j < 822; ++j) s += Ybr[l * 822 + j] * IM[j * 600 + k];
          v = s;
        }
      }
    }
    __bf16 h, lo; split_bf16(v, h, lo);
    Wh[row * WV_K + k] = h; Wl[row * WV_K + k] = lo;
  }
}

__global__ void pack_wp(const float* __restrict__ MapL,
                        __bf16* __restrict__ Wh, __bf16* __restrict__ Wl) {
  const int n = blockIdx.x;
  for (int k = threadIdx.x; k < WP_K; k += 256) {
    float v = (n < 600 && k < 600) ? MapL[k * 600 + n] : 0.f;  // transpose
    __bf16 h, lo; split_bf16(v, h, lo);
    Wh[n * WP_K + k] = h; Wl[n * WP_K + k] = lo;
  }
}

__global__ void pack_wgs(const float* __restrict__ Mapg,
                         __bf16* __restrict__ Wh, __bf16* __restrict__ Wl) {
  const int j = blockIdx.x;
  for (int k = threadIdx.x; k < WGS_K; k += 256) {
    float v = (j < 138 && k < 600) ? Mapg[j * 600 + k] : 0.f;
    __bf16 h, lo; split_bf16(v, h, lo);
    Wh[j * WGS_K + k] = h; Wl[j * WGS_K + k] = lo;
  }
}

__global__ void pack_wgen(const float* __restrict__ Mapg,
                          __bf16* __restrict__ Wh, __bf16* __restrict__ Wl) {
  const int n = blockIdx.x;
  for (int k = threadIdx.x; k < WGEN_K; k += 256) {
    float v = (n < 600 && k < 138) ? Mapg[k * 600 + n] : 0.f;  // transpose
    __bf16 h, lo; split_bf16(v, h, lo);
    Wh[n * WGEN_K + k] = h; Wl[n * WGEN_K + k] = lo;
  }
}

// ---------------- fused WMMA main kernel ----------------
// 8 waves = 2 M-groups x 4 N-groups; each wave owns 2 N-tiles (2 accumulators).
struct Smem {
  __bf16 Ah[32 * A_STRIDE];
  __bf16 Al[32 * A_STRIDE];
  __bf16 Bst[2 * 4 * 2 * 32 * B_ROWP];   // [buf][group][hi/lo][32 n-rows][B_ROWP]
  float  rsum[4][32];
  float  sred[256];
};

union FragU { v16bf v; u32x4 q[2]; };

__device__ __forceinline__ void wait_async0() {
  asm volatile("s_wait_asynccnt 0x0" ::: "memory");
}

// CDNA5 async DMA: global -> LDS, tracked by ASYNCcnt (no VGPR round trip).
// Stage one K-step (32 k) of B panels for all 4 groups x 2 tiles x (hi,lo):
// 1024 x 16B chunks, 4 async b128 ops per thread.
__device__ __forceinline__ void stage_async(Smem& sm, int tid,
                                            const __bf16* __restrict__ Wh,
                                            const __bf16* __restrict__ Wl,
                                            int KW, int tg, int ks, int buf) {
#pragma unroll
  for (int ii = 0; ii < 4; ++ii) {
    const int i   = tid + ii * 256;
    const int q   = i & 3;
    const int row = (i >> 2) & 31;          // 0..31 (2 tiles of 16 n)
    const int hl  = (i >> 7) & 1;
    const int g   = (i >> 8) & 3;
    const int ng  = tg * 128 + g * 32 + row;
    const __bf16* src = (hl ? Wl : Wh) + (size_t)ng * KW + ks * 32 + q * 8;
    __bf16* dst = sm.Bst + ((((buf * 4 + g) * 2 + hl) * 32 + row) * B_ROWP) + q * 8;
    unsigned       ldsOff = (unsigned)(size_t)((char*)dst - (char*)&sm);
    unsigned long long ga = (unsigned long long)src;
    asm volatile("global_load_async_to_lds_b128 %0, %1, off"
                 :: "v"(ldsOff), "v"(ga) : "memory");
  }
}

// type 0: Volt GEMM (quad_p/quad_q -> scalarAcc; Ibr line limits -> rowAcc)
// type 1: +Bf * P_Loads@Map_L  masked col-sum
// type 2: -Bf * P_Gens@Map_g   masked col-sum
// type 3: |stat| with mu_g_u/mu_g_d/cost
__device__ __forceinline__ void run_phase(
    Smem& sm, int tid, long rowBase,
    const float* __restrict__ Aglob, int Kact, int KloadPad,
    const __bf16* __restrict__ Wh, const __bf16* __restrict__ Wl,
    int KW, int KS, int TG, int type,
    const float* __restrict__ mu_i, const float* __restrict__ mgu,
    const float* __restrict__ mgd, const float* __restrict__ cpg,
    float rowAcc[8], float& scalarAcc) {
  const int lane = tid & 31, w = tid >> 5;
  const int g = w >> 2, mt = w & 3;         // placeholder (overwritten below)
  (void)g; (void)mt;
  const int grp = w >> 1, mtile = w & 1;
  const int half = lane >> 4, nl = lane & 15;

  // load + split A tile (32 rows x KloadPad) into LDS; float2 loads (rows 8B aligned)
  for (int c2 = tid; c2 < (KloadPad >> 1); c2 += 256) {
    for (int r = 0; r < 32; ++r) {
      float x0 = 0.f, x1 = 0.f;
      if (c2 * 2 + 1 < Kact) {
        float2 x = *reinterpret_cast<const float2*>(&Aglob[(rowBase + r) * Kact + c2 * 2]);
        x0 = x.x; x1 = x.y;
      }
      __bf16 h0, l0, h1, l1;
      split_bf16(x0, h0, l0); split_bf16(x1, h1, l1);
      sm.Ah[r * A_STRIDE + c2 * 2]     = h0;
      sm.Ah[r * A_STRIDE + c2 * 2 + 1] = h1;
      sm.Al[r * A_STRIDE + c2 * 2]     = l0;
      sm.Al[r * A_STRIDE + c2 * 2 + 1] = l1;
    }
  }
  __syncthreads();

  for (int tg = 0; tg < TG; ++tg) {
    v8f accs[2];
    accs[0] = (v8f){0.f, 0.f, 0.f, 0.f, 0.f, 0.f, 0.f, 0.f};
    accs[1] = (v8f){0.f, 0.f, 0.f, 0.f, 0.f, 0.f, 0.f, 0.f};
    stage_async(sm, tid, Wh, Wl, KW, tg, 0, 0);
    wait_async0();
    __syncthreads();
    for (int ks = 0; ks < KS; ++ks) {
      if (ks + 1 < KS) stage_async(sm, tid, Wh, Wl, KW, tg, ks + 1, (ks + 1) & 1);
      const int buf = ks & 1;
      FragU ah, al;
      const int arow = mtile * 16 + nl;
      const int k8   = ks * 32 + 8 * half;  // A layout: half h holds K {8h..8h+7, 16+8h..}
      const __bf16* ap  = &sm.Ah[arow * A_STRIDE + k8];
      const __bf16* alp = &sm.Al[arow * A_STRIDE + k8];
      ah.q[0] = *reinterpret_cast<const u32x4*>(ap);
      ah.q[1] = *reinterpret_cast<const u32x4*>(ap + 16);
      al.q[0] = *reinterpret_cast<const u32x4*>(alp);
      al.q[1] = *reinterpret_cast<const u32x4*>(alp + 16);
#pragma unroll
      for (int t = 0; t < 2; ++t) {
        FragU bh, bl;
        const __bf16* bp  = &sm.Bst[(((buf * 4 + grp) * 2 + 0) * 32 + (t * 16 + nl)) * B_ROWP + 8 * half];
        const __bf16* blp = bp + 32 * B_ROWP;   // hi/lo plane stride
        bh.q[0] = *reinterpret_cast<const u32x4*>(bp);
        bh.q[1] = *reinterpret_cast<const u32x4*>(bp + 16);
        bl.q[0] = *reinterpret_cast<const u32x4*>(blp);
        bl.q[1] = *reinterpret_cast<const u32x4*>(blp + 16);
        // split-bf16: hi*hi + lo*hi + hi*lo chained into f32 accumulator
        accs[t] = __builtin_amdgcn_wmma_f32_16x16x32_bf16(false, ah.v, false, bh.v, (short)0, accs[t], false, false);
        accs[t] = __builtin_amdgcn_wmma_f32_16x16x32_bf16(false, al.v, false, bh.v, (short)0, accs[t], false, false);
        accs[t] = __builtin_amdgcn_wmma_f32_16x16x32_bf16(false, ah.v, false, bl.v, (short)0, accs[t], false, false);
      }
      wait_async0();      // staged K-step complete before the barrier hand-off
      __syncthreads();
    }

    // ---- fused epilogue: C layout lane n=lane&15, rows m = r + 8*half ----
#pragma unroll
    for (int t = 0; t < 2; ++t) {
      const int nbase = (tg * 8 + grp * 2 + t) * 16 + nl;
      if (type == 0) {
        if (nbase < 1200) {             // quad_p / quad_q -> global scalar
          const int c = (nbase >= 600) ? nbase - 600 : nbase;
          if (c != 299 && c != 599) {
#pragma unroll
            for (int r = 0; r < 8; ++r) {
              const int mrow = mtile * 16 + 8 * half + r;
              float vv = (float)sm.Ah[mrow * A_STRIDE + c] + (float)sm.Al[mrow * A_STRIDE + c];
              scalarAcc += vv * accs[t][r];
            }
          }
        } else {                        // Ibr: (re,im) interleaved pairs
          const int p  = nbase - 1200;
          const int tl = p >> 1;
#pragma unroll
          for (int r = 0; r < 8; ++r) {
            float o  = __shfl_xor(accs[t][r], 1, 32);      // partner of the pair
            float iu = accs[t][r] * accs[t][r] + o * o - 1.0f;
            if (((p & 1) == 0) && (tl < NLINE)) {
              long b = rowBase + mtile * 16 + 8 * half + r;
              float mi = mu_i[b * NLINE + tl];
              rowAcc[r] += fmaxf(iu, 0.f) + fabsf(mi * iu);
            }
          }
        }
      } else if (type == 1 || type == 2) {
        if (nbase < 600 && nbase != 299 && nbase != 599) {
          const float f = (type == 1) ? BATCH_F : -BATCH_F;
#pragma unroll
          for (int r = 0; r < 8; ++r) rowAcc[r] += f * accs[t][r];
        }
      } else {                          // stat
        if (nbase < N2G) {
          const float cost = (nbase < 69) ? cpg[nbase] : 0.f;
#pragma unroll
          for (int r = 0; r < 8; ++r) {
            long b = rowBase + mtile * 16 + 8 * half + r;
            float st = accs[t][r] + mgu[b * N2G + nbase] - mgd[b * N2G + nbase] - cost;
            rowAcc[r] += fabsf(st);
          }
        }
      }
    }
  }
  __syncthreads();   // A region reused by next phase
}

__global__ void __launch_bounds__(256) main_kernel(
    const float* __restrict__ Volt, const float* __restrict__ P_Gens,
    const float* __restrict__ P_Loads, const float* __restrict__ nolp,
    const float* __restrict__ mgu, const float* __restrict__ mgd,
    const float* __restrict__ mu_i, const float* __restrict__ cpg,
    const __bf16* __restrict__ WvH, const __bf16* __restrict__ WvL,
    const __bf16* __restrict__ WpH, const __bf16* __restrict__ WpL,
    const __bf16* __restrict__ WgsH, const __bf16* __restrict__ WgsL,
    const __bf16* __restrict__ WgenH, const __bf16* __restrict__ WgenL,
    float* __restrict__ out, float* __restrict__ part) {
  __shared__ Smem sm;
  const int tid = threadIdx.x;
  const int lane = tid & 31, w = tid >> 5;
  const int grp = w >> 1, mtile = w & 1;
  const int half = lane >> 4;
  const long rowBase = (long)blockIdx.x * 32;

  float rowAcc[8] = {0.f, 0.f, 0.f, 0.f, 0.f, 0.f, 0.f, 0.f};
  float scalarAcc = 0.f;
  if (tid < 32) scalarAcc += fabsf(Volt[(rowBase + tid) * N2BUS + 300]);  // kkt0 partial

  run_phase(sm, tid, rowBase, Volt,    600, 608, WvH,   WvL,   WV_K,   19, 16, 0,
            mu_i, nullptr, nullptr, nullptr, rowAcc, scalarAcc);
  run_phase(sm, tid, rowBase, P_Loads, 600, 608, WpH,   WpL,   WP_K,   19,  5, 1,
            nullptr, nullptr, nullptr, nullptr, rowAcc, scalarAcc);
  run_phase(sm, tid, rowBase, P_Gens,  138, 160, WgenH, WgenL, WGEN_K,  5,  5, 2,
            nullptr, nullptr, nullptr, nullptr, rowAcc, scalarAcc);
  run_phase(sm, tid, rowBase, nolp,    600, 608, WgsH,  WgsL,  WGS_K,  19,  2, 3,
            nullptr, mgu, mgd, cpg, rowAcc, scalarAcc);

  // reduce per-row terms across the 16 lanes of each half, then across the 4 n-groups
#pragma unroll
  for (int r = 0; r < 8; ++r) {
    float v = rowAcc[r];
#pragma unroll
    for (int off = 1; off < 16; off <<= 1) v += __shfl_xor(v, off, 32);
    rowAcc[r] = v;
  }
  if ((lane & 15) == 0) {
#pragma unroll
    for (int r = 0; r < 8; ++r) sm.rsum[grp][mtile * 16 + 8 * half + r] = rowAcc[r];
  }
  sm.sred[tid] = scalarAcc;
  __syncthreads();
  if (tid < 32)
    out[rowBase + tid] = sm.rsum[0][tid] + sm.rsum[1][tid] + sm.rsum[2][tid] + sm.rsum[3][tid];
  for (int st = 128; st > 0; st >>= 1) {
    if (tid < st) sm.sred[tid] += sm.sred[tid + st];
    __syncthreads();
  }
  if (tid == 0) part[blockIdx.x] = sm.sred[0];
}

// ---------------- finalize: deterministic scalar broadcast + elementwise terms ----------------
__global__ void __launch_bounds__(256) finalize_kernel(
    const float* __restrict__ part,
    const float* __restrict__ Volt, const float* __restrict__ P_Gens,
    const float* __restrict__ mgu, const float* __restrict__ mgd,
    const float* __restrict__ mvu, const float* __restrict__ mvd,
    const float* __restrict__ mui,
    const float* __restrict__ Gen_max, const float* __restrict__ Gen_min,
    float* __restrict__ out) {
  __shared__ float sred[256];
  const int tid = threadIdx.x;
  float s = 0.f;
  for (int i = tid; i < NUM_WG; i += 256) s += part[i];
  sred[tid] = s; __syncthreads();
  for (int st = 128; st > 0; st >>= 1) {
    if (tid < st) sred[tid] += sred[tid + st];
    __syncthreads();
  }
  const float S = sred[0];   // scalar_pq + kkt0, identical (bitwise) in every block

  const int w = tid >> 5, lane = tid & 31;
  const long b = (long)blockIdx.x * 8 + w;
  float t = 0.f;
  for (int j = lane; j < N2G; j += 32) {
    float pg = P_Gens[b * N2G + j];
    float d1 = pg - Gen_max[j], d2 = Gen_min[j] - pg;
    float m1 = mgu[b * N2G + j], m2 = mgd[b * N2G + j];
    t += fmaxf(d1, 0.f) + fmaxf(d2, 0.f)
       + (fabsf(m1 * d1) + fabsf(m2 * d2)) * (1.0f / 69.0f)
       + fmaxf(-m1, 0.f) + fmaxf(-m2, 0.f);
  }
  for (int c = lane; c < 300; c += 32) {
    float vr = Volt[b * N2BUS + c], vi = Volt[b * N2BUS + 300 + c];
    float vu = vr * vr + vi * vi - 1.06f * 1.06f;
    float vd = 0.94f * 0.94f - vr * vr + vi * vi;   // matches reference (sic: +vi^2)
    float m1 = mvu[b * 300 + c], m2 = mvd[b * 300 + c];
    t += fmaxf(vu, 0.f) + fmaxf(vd, 0.f) + fabsf(m1 * vu) + fabsf(m2 * vd)
       + fmaxf(-m1, 0.f) + fmaxf(-m2, 0.f);
  }
  for (int l = lane; l < NLINE; l += 32) t += fmaxf(-mui[b * NLINE + l], 0.f);
#pragma unroll
  for (int off = 16; off > 0; off >>= 1) t += __shfl_xor(t, off, 32);
  if (lane == 0) out[b] += S + t;
}

// ---------------- host launch ----------------
extern "C" void kernel_launch(void* const* d_in, const int* in_sizes, int n_in,
                              void* d_out, int out_size, void* d_ws, size_t ws_size,
                              hipStream_t stream) {
  (void)in_sizes; (void)n_in; (void)out_size; (void)ws_size;
  const float* Volt    = (const float*)d_in[0];
  const float* P_Gens  = (const float*)d_in[1];
  const float* P_Loads = (const float*)d_in[2];
  const float* nolp    = (const float*)d_in[3];
  const float* mgu     = (const float*)d_in[4];
  const float* mgd     = (const float*)d_in[5];
  const float* mvu     = (const float*)d_in[6];
  const float* mvd     = (const float*)d_in[7];
  const float* mui     = (const float*)d_in[8];
  const float* Y       = (const float*)d_in[9];
  const float* Yconj   = (const float*)d_in[10];
  const float* Ybr     = (const float*)d_in[11];
  const float* IM      = (const float*)d_in[12];
  const float* Mapg    = (const float*)d_in[13];
  const float* MapL    = (const float*)d_in[14];
  const float* Gen_max = (const float*)d_in[15];
  const float* Gen_min = (const float*)d_in[16];
  const float* C_Pg    = (const float*)d_in[17];
  float* out = (float*)d_out;

  char* ws = (char*)d_ws;
  size_t off = 0;
  __bf16* WvH   = (__bf16*)(ws + off); off += (size_t)WV_ROWS   * WV_K   * 2;
  __bf16* WvL   = (__bf16*)(ws + off); off += (size_t)WV_ROWS   * WV_K   * 2;
  __bf16* WpH   = (__bf16*)(ws + off); off += (size_t)WP_ROWS   * WP_K   * 2;
  __bf16* WpL   = (__bf16*)(ws + off); off += (size_t)WP_ROWS   * WP_K   * 2;
  __bf16* WgsH  = (__bf16*)(ws + off); off += (size_t)WGS_ROWS  * WGS_K  * 2;
  __bf16* WgsL  = (__bf16*)(ws + off); off += (size_t)WGS_ROWS  * WGS_K  * 2;
  __bf16* WgenH = (__bf16*)(ws + off); off += (size_t)WGEN_ROWS * WGEN_K * 2;
  __bf16* WgenL = (__bf16*)(ws + off); off += (size_t)WGEN_ROWS * WGEN_K * 2;
  float*  part  = (float*)(ws + off);  off += (size_t)NUM_WG * 4;

  pack_wv  <<<WV_ROWS,   256, 0, stream>>>(Y, Yconj, Ybr, IM, WvH, WvL);
  pack_wp  <<<WP_ROWS,   256, 0, stream>>>(MapL, WpH, WpL);
  pack_wgs <<<WGS_ROWS,  256, 0, stream>>>(Mapg, WgsH, WgsL);
  pack_wgen<<<WGEN_ROWS, 256, 0, stream>>>(Mapg, WgenH, WgenL);

  main_kernel<<<NUM_WG, 256, 0, stream>>>(
      Volt, P_Gens, P_Loads, nolp, mgu, mgd, mui, C_Pg,
      WvH, WvL, WpH, WpL, WgsH, WgsL, WgenH, WgenL, out, part);

  finalize_kernel<<<BATCH_N / 8, 256, 0, stream>>>(
      part, Volt, P_Gens, mgu, mgd, mvu, mvd, mui, Gen_max, Gen_min, out);
}
